// DeepseekCompressor_51359218925904
// MI455X (gfx1250) — matrix operand order repaired and checked by reading, not compiled, not run on hardware
//
#include <hip/hip_runtime.h>

// ---------------------------------------------------------------------------
// DeepseekCompressor: out = scatter(concat(x@W.T[:,:512],
//                             x@W.T[:,512:] + ape[pos%128]), slot_mapping)
// bf16x3 split-precision GEMM on v_wmma_f32_16x16x32_bf16 (wave32).
// Phase 1 (if workspace allows): split x,W into bf16 hi/lo arrays once.
// Phase 2: block tile 128x128, BK=64, 8 waves of 32x64, f32 accumulate.
// ---------------------------------------------------------------------------

typedef __attribute__((ext_vector_type(16))) __bf16 v16bf;
typedef __attribute__((ext_vector_type(8)))  float  v8f;

static constexpr int KDIM   = 7168;   // hidden
static constexpr int NDIM   = 1024;   // 2 * coff * head_dim
static constexpr int DHALF  = 512;    // coff * head_dim
static constexpr int CRATIO = 128;    // compress_ratio (ape rows)
static constexpr int BM     = 128;
static constexpr int BN     = 128;
static constexpr int BK     = 64;
static constexpr int LDP    = 72;     // padded LDS pitch (144B stride -> conflict-free)

__device__ __forceinline__ void bf16_split(float f, unsigned short& hi, unsigned short& lo) {
    unsigned u  = __float_as_uint(f);
    unsigned r  = (u + 0x7FFFu + ((u >> 16) & 1u)) >> 16;   // RNE hi
    hi          = (unsigned short)r;
    float hif   = __uint_as_float(r << 16);
    float res   = f - hif;
    lo          = (unsigned short)(__float_as_uint(res) >> 16);  // truncate lo (tiny err)
}

union FragCast { v16bf v; uint4 u[2]; };

// A fragment (16x32 bf16): lane m=lane&15, half=lane>>4.
__device__ __forceinline__ v16bf load_fragA(const unsigned short* base, int row16,
                                            int koff, int lane) {
    const int m    = lane & 15;
    const int half = lane >> 4;
    const unsigned short* p = base + (row16 + m) * LDP + koff + half * 8;
    FragCast f;
    f.u[0] = *reinterpret_cast<const uint4*>(p);
    f.u[1] = *reinterpret_cast<const uint4*>(p + 16);
    return f.v;
}

// B fragment (32x16 bf16, col n=lane&15): lanes 0-15 hold K=0..7,16..23, 16-31 K=8..15,24..31
__device__ __forceinline__ v16bf load_fragB(const unsigned short* base, int col16,
                                            int koff, int lane) {
    const int n    = lane & 15;
    const int half = lane >> 4;
    const unsigned short* p = base + (col16 + n) * LDP + koff + half * 8;
    FragCast f;
    f.u[0] = *reinterpret_cast<const uint4*>(p);
    f.u[1] = *reinterpret_cast<const uint4*>(p + 16);
    return f.v;
}

// -------- phase 1: one-shot f32 -> (bf16 hi, bf16 lo) split ---------------
__global__ __launch_bounds__(256) void split_kernel(const float4* __restrict__ src,
                                                    ushort4* __restrict__ hi,
                                                    ushort4* __restrict__ lo,
                                                    long n4) {
    const long idx = (long)blockIdx.x * 256 + threadIdx.x;
    if (idx < n4) {
        const float4 v = src[idx];
        ushort4 hv, lv;
        bf16_split(v.x, hv.x, lv.x);
        bf16_split(v.y, hv.y, lv.y);
        bf16_split(v.z, hv.z, lv.z);
        bf16_split(v.w, hv.w, lv.w);
        hi[idx] = hv;
        lo[idx] = lv;
    }
}

// -------- phase 2: GEMM + epilogue + scatter ------------------------------
template <bool PRE>
__global__ __launch_bounds__(256) void dsc_gemm_scatter(
    const float* __restrict__ x,   const unsigned short* __restrict__ Xh,
    const unsigned short* __restrict__ Xl,
    const float* __restrict__ W,   const unsigned short* __restrict__ Wh,
    const unsigned short* __restrict__ Wl,
    const float* __restrict__ ape,
    const int*   __restrict__ positions,
    const int*   __restrict__ slot_mapping,
    float*       __restrict__ out,
    int numSlots)
{
    __shared__ unsigned short Ah[BM * LDP];
    __shared__ unsigned short Al[BM * LDP];
    __shared__ unsigned short Bh[BN * LDP];
    __shared__ unsigned short Bl[BN * LDP];
    __shared__ int smap[BM];
    __shared__ int spos[BM];

    const int tid   = threadIdx.x;
    const int lane  = tid & 31;
    const int wid   = tid >> 5;
    const int waveM = (wid >> 1) * 32;           // 4 waves along M
    const int waveN = (wid & 1) * 64;            // 2 waves along N
    const int rowBase = blockIdx.y * BM;
    const int colBase = blockIdx.x * BN;

    if (tid < BM) {
        smap[tid] = slot_mapping[rowBase + tid];
        spos[tid] = positions[rowBase + tid];
    }

    v8f acc[2][4] = {};

    for (int kk = 0; kk < KDIM; kk += BK) {
        __syncthreads();

        if (PRE) {
            // 128x64 bf16 tiles: 1024 chunks of 8 bf16 -> 4 uint4 per thread/array
#pragma unroll
            for (int i = 0; i < 4; ++i) {
                const int idx = tid + i * 256;
                const int row = idx >> 3;
                const int c8  = idx & 7;
                const size_t gA = (size_t)(rowBase + row) * KDIM + kk + c8 * 8;
                const size_t gB = (size_t)(colBase + row) * KDIM + kk + c8 * 8;
                if (kk + BK < KDIM) __builtin_prefetch(Xh + gA + BK, 0, 1);
                *reinterpret_cast<uint4*>(&Ah[row * LDP + c8 * 8]) =
                    *reinterpret_cast<const uint4*>(Xh + gA);
                *reinterpret_cast<uint4*>(&Al[row * LDP + c8 * 8]) =
                    *reinterpret_cast<const uint4*>(Xl + gA);
                *reinterpret_cast<uint4*>(&Bh[row * LDP + c8 * 8]) =
                    *reinterpret_cast<const uint4*>(Wh + gB);
                *reinterpret_cast<uint4*>(&Bl[row * LDP + c8 * 8]) =
                    *reinterpret_cast<const uint4*>(Wl + gB);
            }
        } else {
            // convert in loop: 128x64 f32 tiles = 2048 float4 chunks
#pragma unroll
            for (int i = 0; i < 8; ++i) {
                const int idx = tid + i * 256;
                const int row = idx >> 4;
                const int c4  = idx & 15;
                const float* gp = x + (size_t)(rowBase + row) * KDIM + kk + c4 * 4;
                if (kk + BK < KDIM) __builtin_prefetch(gp + BK, 0, 1);
                const float4 v = *reinterpret_cast<const float4*>(gp);
                ushort4 hv, lv;
                bf16_split(v.x, hv.x, lv.x);
                bf16_split(v.y, hv.y, lv.y);
                bf16_split(v.z, hv.z, lv.z);
                bf16_split(v.w, hv.w, lv.w);
                *reinterpret_cast<ushort4*>(&Ah[row * LDP + c4 * 4]) = hv;
                *reinterpret_cast<ushort4*>(&Al[row * LDP + c4 * 4]) = lv;
            }
#pragma unroll
            for (int i = 0; i < 8; ++i) {
                const int idx = tid + i * 256;
                const int row = idx >> 4;
                const int c4  = idx & 15;
                const float4 v = *reinterpret_cast<const float4*>(
                    W + (size_t)(colBase + row) * KDIM + kk + c4 * 4);
                ushort4 hv, lv;
                bf16_split(v.x, hv.x, lv.x);
                bf16_split(v.y, hv.y, lv.y);
                bf16_split(v.z, hv.z, lv.z);
                bf16_split(v.w, hv.w, lv.w);
                *reinterpret_cast<ushort4*>(&Bh[row * LDP + c4 * 4]) = hv;
                *reinterpret_cast<ushort4*>(&Bl[row * LDP + c4 * 4]) = lv;
            }
        }
        __syncthreads();

#pragma unroll
        for (int s = 0; s < 2; ++s) {            // two K=32 sub-steps
            const int koff = s * 32;
            v16bf aH[2], aL[2], bH[4], bL[4];
#pragma unroll
            for (int mt = 0; mt < 2; ++mt) {
                aH[mt] = load_fragA(Ah, waveM + mt * 16, koff, lane);
                aL[mt] = load_fragA(Al, waveM + mt * 16, koff, lane);
            }
#pragma unroll
            for (int nt = 0; nt < 4; ++nt) {
                bH[nt] = load_fragB(Bh, waveN + nt * 16, koff, lane);
                bL[nt] = load_fragB(Bl, waveN + nt * 16, koff, lane);
            }
#pragma unroll
            for (int mt = 0; mt < 2; ++mt) {
#pragma unroll
                for (int nt = 0; nt < 4; ++nt) {
                    acc[mt][nt] = __builtin_amdgcn_wmma_f32_16x16x32_bf16(
                        false, aH[mt], false, bH[nt], (short)0, acc[mt][nt], false, false);
                    acc[mt][nt] = __builtin_amdgcn_wmma_f32_16x16x32_bf16(
                        false, aH[mt], false, bL[nt], (short)0, acc[mt][nt], false, false);
                    acc[mt][nt] = __builtin_amdgcn_wmma_f32_16x16x32_bf16(
                        false, aL[mt], false, bH[nt], (short)0, acc[mt][nt], false, false);
                }
            }
        }
    }
    __syncthreads();

    // epilogue: ape add (uniform per block) + row scatter
    const bool isScore = (colBase >= DHALF);
    const int half = lane >> 4;
    const int nloc = lane & 15;
#pragma unroll
    for (int mt = 0; mt < 2; ++mt) {
#pragma unroll
        for (int nt = 0; nt < 4; ++nt) {
            const int col = colBase + waveN + nt * 16 + nloc;
#pragma unroll
            for (int v = 0; v < 8; ++v) {
                const int mloc = waveM + mt * 16 + v + 8 * half;
                float val = acc[mt][nt][v];
                if (isScore) {
                    const int pos = spos[mloc];
                    val += ape[(size_t)(pos & (CRATIO - 1)) * DHALF + (col - DHALF)];
                }
                const int slot = smap[mloc];
                if (slot >= 0 && slot < numSlots)
                    out[(size_t)slot * NDIM + col] = val;
            }
        }
    }
}

extern "C" void kernel_launch(void* const* d_in, const int* in_sizes, int n_in,
                              void* d_out, int out_size, void* d_ws, size_t ws_size,
                              hipStream_t stream) {
    const float* x            = (const float*)d_in[0];
    const float* W            = (const float*)d_in[1];
    const float* ape          = (const float*)d_in[2];
    const float* state_cache  = (const float*)d_in[3];
    const int*   positions    = (const int*)d_in[4];
    const int*   slot_mapping = (const int*)d_in[5];

    const int M        = in_sizes[4];
    const int numSlots = in_sizes[3] / NDIM;

    // init output from input state_cache (reference scatters into it)
    hipMemcpyAsync(d_out, state_cache, (size_t)out_size * sizeof(float),
                   hipMemcpyDeviceToDevice, stream);

    const size_t nX = (size_t)M * KDIM;            // x elements
    const size_t nW = (size_t)NDIM * KDIM;         // W elements
    const size_t need = (nX + nW) * 2 * sizeof(unsigned short); // hi+lo bf16

    dim3 grid(NDIM / BN, M / BM);                  // N fastest -> x reuse in L2

    if (ws_size >= need) {
        unsigned short* Xh = (unsigned short*)d_ws;
        unsigned short* Xl = Xh + nX;
        unsigned short* Wh = Xl + nX;
        unsigned short* Wl = Wh + nW;
        const long nX4 = (long)(nX / 4), nW4 = (long)(nW / 4);
        split_kernel<<<(unsigned)((nX4 + 255) / 256), 256, 0, stream>>>(
            (const float4*)x, (ushort4*)Xh, (ushort4*)Xl, nX4);
        split_kernel<<<(unsigned)((nW4 + 255) / 256), 256, 0, stream>>>(
            (const float4*)W, (ushort4*)Wh, (ushort4*)Wl, nW4);
        dsc_gemm_scatter<true><<<grid, 256, 0, stream>>>(
            nullptr, Xh, Xl, nullptr, Wh, Wl, ape, positions, slot_mapping,
            (float*)d_out, numSlots);
    } else {
        dsc_gemm_scatter<false><<<grid, 256, 0, stream>>>(
            x, nullptr, nullptr, W, nullptr, nullptr, ape, positions, slot_mapping,
            (float*)d_out, numSlots);
    }
}
